// DiscrimModel_50474455663060
// MI455X (gfx1250) — matrix operand order repaired and checked by reading, not compiled, not run on hardware
//
#include <hip/hip_runtime.h>
#include <stdint.h>

typedef __attribute__((ext_vector_type(16))) __bf16 v16bf;
typedef __attribute__((ext_vector_type(8)))  float  v8f;
typedef __attribute__((ext_vector_type(4)))  unsigned int u32x4;
typedef __attribute__((ext_vector_type(8)))  int i32x8;
typedef __attribute__((ext_vector_type(4)))  int i32x4;

__device__ __forceinline__ uint16_t f2bf(float x) {
  uint32_t u = __float_as_uint(x);
  u += 0x7FFFu + ((u >> 16) & 1u);   // round-to-nearest-even
  return (uint16_t)(u >> 16);
}
__device__ __forceinline__ float bf2f(uint16_t b) {
  return __uint_as_float(((uint32_t)b) << 16);
}

#define FD 64
#define FH 384
#define FW 384
#define PD 16
#define PH 128
#define PW 128
#define NPATCH 8

// ---------------- zero output ----------------
__global__ void zero_f(float* __restrict__ p, int n) {
  int i = blockIdx.x * blockDim.x + threadIdx.x;
  if (i < n) p[i] = 0.f;
}

// ---------------- build bf16 [ml, img] input ----------------
__global__ void prep(const float* __restrict__ img, const int* __restrict__ lab,
                     const int* __restrict__ focus, uint16_t* __restrict__ x0) {
  const int per = PD * PH * PW;
  int tid = blockIdx.x * blockDim.x + threadIdx.x;
  if (tid >= NPATCH * per) return;
  int p = tid / per;
  int r = tid % per;
  int w = r % PW;
  int h = (r / PW) % PH;
  int d = r / (PW * PH);
  int f0 = focus[p * 3 + 0], f1 = focus[p * 3 + 1], f2 = focus[p * 3 + 2];
  size_t gl = ((size_t)(f0 + d) * FH + (f1 + h)) * FW + (f2 + w);
  size_t gc = ((size_t)(f0 + PD / 2) * FH + (f1 + PH / 2)) * FW + (f2 + PW / 2);
  float ml = (lab[gl] == lab[gc]) ? 1.f : 0.f;
  x0[(size_t)tid * 2 + 0] = f2bf(ml);
  x0[(size_t)tid * 2 + 1] = f2bf(img[gl]);
}

// ---------------- pack weights into B-fragment layout ----------------
// layout: [ntile][ktile][lane 0..31][vgpr 0..7] dwords -> lane's 8 dwords contiguous (2x ds_load_b128)
// B 32x16 bf16: lane holds col n = 16*ntile + (lane&15);
// lanes 0-15 -> K base 0, lanes 16-31 -> K base 16; VGPR v -> k = base + 2v, 2v+1
template <int CIN, int COUT>
__global__ void pack_w(const float* __restrict__ W, uint32_t* __restrict__ pw) {
  constexpr int K  = 27 * CIN;
  constexpr int KT = (K + 31) / 32;
  constexpr int NT = COUT / 16;
  int tid = blockIdx.x * blockDim.x + threadIdx.x;
  if (tid >= NT * KT * 32 * 8) return;
  int v    = tid & 7;
  int lane = (tid >> 3) & 31;
  int kt   = (tid >> 8) % KT;
  int nt   = tid / (KT * 256);
  int n  = nt * 16 + (lane & 15);
  int k0 = kt * 32 + ((lane >> 4) ? 16 : 0) + 2 * v;
  uint32_t lo = (k0     < K) ? (uint32_t)f2bf(W[(size_t)k0 * COUT + n])       : 0u;
  uint32_t hi = (k0 + 1 < K) ? (uint32_t)f2bf(W[(size_t)(k0 + 1) * COUT + n]) : 0u;
  pw[tid] = lo | (hi << 16);
}

// ---- compile-time im2col helpers (fold to literals once the KT loop is unrolled) ----
__device__ __forceinline__ constexpr int im2col_off(int k0, int CIN, int IH, int IW) {
  const int tap = k0 / CIN, c = k0 % CIN;
  const int kd = tap / 9, kh = (tap / 3) % 3, kw = tap % 3;
  return ((kd * IH + kh) * IW + kw) * CIN + c;
}
__device__ __forceinline__ bool im2col_ok(int k0, int CIN, bool okD2, bool okH2, bool okW2) {
  const int tap = k0 / CIN;
  const int kd = tap / 9, kh = (tap / 3) % 3, kw = tap % 3;
  return (kd < 2 || okD2) && (kh < 2 || okH2) && (kw < 2 || okW2);
}

// ---- TDM: DMA one contiguous panel of `ndw` dwords from global into LDS ----
__device__ __forceinline__ void tdm_load_panel(const uint32_t* __restrict__ gsrc,
                                               uint32_t* ldsDst, int ndw) {
#if __has_builtin(__builtin_amdgcn_tensor_load_to_lds) && __has_builtin(__builtin_amdgcn_s_wait_tensorcnt)
  const uint64_t ga = (uint64_t)(uintptr_t)gsrc;
  const uint32_t la =
      (uint32_t)(uintptr_t)(__attribute__((address_space(3))) uint32_t*)ldsDst;
  u32x4 g0;
  g0[0] = 1u;                                                // count=1, user desc
  g0[1] = la;                                                // lds_addr (bytes)
  g0[2] = (uint32_t)ga;                                      // global_addr[31:0]
  g0[3] = (uint32_t)((ga >> 32) & 0x1FFFFFFu) | (2u << 30);  // global_addr[56:32] | type=2
  i32x8 g1;
  g1[0] = (int)(2u << 16);                                   // wg_mask=0, data_size=4B
  g1[1] = (int)(((uint32_t)ndw & 0xFFFFu) << 16);            // tensor_dim0[15:0]
  g1[2] = (int)(((uint32_t)ndw >> 16) | (1u << 16));         // tensor_dim0[31:16], tensor_dim1=1
  g1[3] = (int)((uint32_t)ndw << 16);                        // tile_dim0 = ndw
  g1[4] = 1;                                                 // tile_dim1 = 1
  g1[5] = ndw;                                               // tensor_dim0_stride[31:0]
  g1[6] = 0;
  g1[7] = 0;
  const i32x4 gz4 = {0, 0, 0, 0};
  const i32x8 gz8 = {0, 0, 0, 0, 0, 0, 0, 0};
  __builtin_amdgcn_tensor_load_to_lds(g0, g1, gz4, gz4, gz8, 0);
  __builtin_amdgcn_s_wait_tensorcnt(0);
#else
  for (int i = (int)(threadIdx.x & 31); i < ndw; i += 32) ldsDst[i] = gsrc[i];
#endif
}

// ---------------- implicit-GEMM conv, stride 2, 3x3x3, SAME (hi-pad only) ----------------
// 4 waves / block, one 16-row M tile per wave; weights TDM'd into LDS; bf16 WMMA, fp32 accum.
template <int ID, int IH, int IW, int CIN, int COUT>
__global__ __launch_bounds__(128) void conv_wmma(
    const uint16_t* __restrict__ xin,   // [NPATCH][ID][IH][IW][CIN] bf16
    const uint32_t* __restrict__ pw,    // [NT][KT][32][8] packed B
    const float*    __restrict__ bias,  // [COUT]
    uint16_t*       __restrict__ xout)  // [NPATCH][OD][OH][OW][COUT] bf16
{
  constexpr int OD = ID / 2, OH = IH / 2, OW = IW / 2;
  constexpr int K   = 27 * CIN;
  constexpr int KT  = (K + 31) / 32;
  constexpr int NDW = KT * 256;   // dwords in this ntile's B panel

  extern __shared__ __align__(16) uint32_t ldsB[];

  const int lane  = threadIdx.x & 31;
  const int wave  = threadIdx.x >> 5;
  const int mtile = blockIdx.x * 4 + wave;
  const int ntile = blockIdx.y;
  const int p     = blockIdx.z;

  if (wave == 0) tdm_load_panel(pw + (size_t)ntile * NDW, ldsB, NDW);
  __syncthreads();

  const int row  = lane & 15;
  const int half = lane >> 4;

  const int s = mtile * 16 + row;
  const int w = s % OW;
  const int h = (s / OW) % OH;
  const int d = s / (OW * OH);

  // stride-2 / k3 SAME: only the hi side pads; OOB possible only for tap index 2
  const bool okD2 = (d != OD - 1);
  const bool okH2 = (h != OH - 1);
  const bool okW2 = (w != OW - 1);

  const uint16_t* base = xin + (size_t)p * ID * IH * IW * CIN +
                         ((size_t)(2 * d) * IH + (2 * h)) * IW * CIN +
                         (size_t)(2 * w) * CIN;

  union Frag { uint32_t u[8]; v16bf v; };
  v8f acc = {};

#pragma unroll
  for (int kt = 0; kt < KT; ++kt) {
    Frag a, b;
    // B fragment: 2x 16B contiguous LDS reads per lane
    const uint32_t* lp = ldsB + ((size_t)(kt * 32 + lane)) * 8;
    const uint4 bA = *(const uint4*)lp;
    const uint4 bB = *(const uint4*)(lp + 4);
    b.u[0] = bA.x; b.u[1] = bA.y; b.u[2] = bA.z; b.u[3] = bA.w;
    b.u[4] = bB.x; b.u[5] = bB.y; b.u[6] = bB.z; b.u[7] = bB.w;

    // A fragment: contiguous k order k = 16*half + 2v, 2v+1 (matches B); both lane-half variants are
    // compile-time constants after unrolling -> select offset, one load.
#pragma unroll
    for (int v = 0; v < 8; ++v) {
      const int k0a = kt * 32 + 2 * v;                        // half = 0
      const int k0b = k0a + 16;                               // half = 1
      const int  offA = im2col_off(k0a, CIN, IH, IW);
      const int  offB = im2col_off(k0b, CIN, IH, IW);
      const bool okA = (k0a < K) && im2col_ok(k0a, CIN, okD2, okH2, okW2);
      const bool okB = (k0b < K) && im2col_ok(k0b, CIN, okD2, okH2, okW2);
      const int  off = half ? offB : offA;
      const bool ok  = half ? okB : okA;
      uint32_t val = 0u;
      if (ok) val = *(const uint32_t*)(base + off);
      a.u[v] = val;
    }
    acc = __builtin_amdgcn_wmma_f32_16x16x32_bf16(false, a.v, false, b.v,
                                                  (short)0, acc, false, false);
  }

  // D layout: lanes 0-15 VGPR r -> M=r ; lanes 16-31 VGPR r -> M=r+8 ; N = lane&15
  const int   n  = ntile * 16 + (lane & 15);
  const float bn = bias[n];
  uint16_t* out = xout + (size_t)p * OD * OH * OW * COUT;
#pragma unroll
  for (int r = 0; r < 8; ++r) {
    const int m  = r + half * 8;
    const int so = mtile * 16 + m;
    float v = acc[r] + bn;
    v = v > 0.f ? v : 0.f;   // ReLU
    out[(size_t)so * COUT + n] = f2bf(v);
  }
}

// ---------------- 1x1x1 logit head + sigmoid ----------------
__global__ void head(const uint16_t* __restrict__ a3, const float* __restrict__ Wout,
                     const float* __restrict__ bout, float* __restrict__ sig) {
  int tid = blockIdx.x * blockDim.x + threadIdx.x;
  if (tid >= NPATCH * 64) return;
  const uint16_t* src = a3 + (size_t)tid * 64;
  float acc = bout[0];
#pragma unroll 8
  for (int c = 0; c < 64; ++c) acc += bf2f(src[c]) * Wout[c];
  sig[tid] = 1.f / (1.f + __expf(-acc));
}

// ---------------- coverage * ml * upsampled-sigmoid, scatter-max ----------------
__global__ void scatter(const int* __restrict__ lab, const int* __restrict__ focus,
                        const float* __restrict__ sig, float* __restrict__ ret) {
  const int per = PD * PH * PW;
  int tid = blockIdx.x * blockDim.x + threadIdx.x;
  if (tid >= NPATCH * per) return;
  int p = tid / per;
  int r = tid % per;
  int w = r % PW;
  int h = (r / PW) % PH;
  int d = r / (PW * PH);
  if (!(d >= 2 && d < 14 && h >= 16 && h < 112 && w >= 16 && w < 112)) return;
  int f0 = focus[p * 3 + 0], f1 = focus[p * 3 + 1], f2 = focus[p * 3 + 2];
  size_t gl = ((size_t)(f0 + d) * FH + (f1 + h)) * FW + (f2 + w);
  size_t gc = ((size_t)(f0 + PD / 2) * FH + (f1 + PH / 2)) * FW + (f2 + PW / 2);
  if (lab[gl] != lab[gc]) return;
  float v = sig[p * 64 + (h >> 4) * 8 + (w >> 4)];
  atomicMax((int*)(ret + gl), __float_as_int(v));   // valid: v >= 0, ret zero-init
}

extern "C" void kernel_launch(void* const* d_in, const int* in_sizes, int n_in,
                              void* d_out, int out_size, void* d_ws, size_t ws_size,
                              hipStream_t stream) {
  const float* img   = (const float*)d_in[0];
  const int*   lab   = (const int*)  d_in[1];
  const int*   focus = (const int*)  d_in[2];
  const float* W0 = (const float*)d_in[3];  const float* b0 = (const float*)d_in[4];
  const float* W1 = (const float*)d_in[5];  const float* b1 = (const float*)d_in[6];
  const float* W2 = (const float*)d_in[7];  const float* b2 = (const float*)d_in[8];
  const float* W3 = (const float*)d_in[9];  const float* b3 = (const float*)d_in[10];
  const float* Wo = (const float*)d_in[11]; const float* bo = (const float*)d_in[12];
  float* ret = (float*)d_out;

  size_t o = 0;
  auto carve = [&](size_t bytes) { size_t r = o; o += (bytes + 255) & ~(size_t)255; return r; };
  char* ws = (char*)d_ws;
  uint16_t* x0 = (uint16_t*)(ws + carve((size_t)NPATCH * PD * PH * PW * 2 * 2));
  uint16_t* a0 = (uint16_t*)(ws + carve((size_t)NPATCH * 8 * 64 * 64 * 16 * 2));
  uint16_t* a1 = (uint16_t*)(ws + carve((size_t)NPATCH * 4 * 32 * 32 * 32 * 2));
  uint16_t* a2 = (uint16_t*)(ws + carve((size_t)NPATCH * 2 * 16 * 16 * 48 * 2));
  uint16_t* a3 = (uint16_t*)(ws + carve((size_t)NPATCH * 1 * 8 * 8 * 64 * 2));
  uint32_t* pw0 = (uint32_t*)(ws + carve((size_t)1 * 2  * 256 * 4));
  uint32_t* pw1 = (uint32_t*)(ws + carve((size_t)2 * 14 * 256 * 4));
  uint32_t* pw2 = (uint32_t*)(ws + carve((size_t)3 * 27 * 256 * 4));
  uint32_t* pw3 = (uint32_t*)(ws + carve((size_t)4 * 41 * 256 * 4));
  float*    sig = (float*)   (ws + carve((size_t)NPATCH * 64 * 4));

  const int outN = FD * FH * FW;
  zero_f<<<(outN + 255) / 256, 256, 0, stream>>>(ret, outN);

  const int perPatch = PD * PH * PW;
  prep<<<(NPATCH * perPatch + 255) / 256, 256, 0, stream>>>(img, lab, focus, x0);

  pack_w<2, 16><<<(1 * 2  * 256 + 255) / 256, 256, 0, stream>>>(W0, pw0);
  pack_w<16, 32><<<(2 * 14 * 256 + 255) / 256, 256, 0, stream>>>(W1, pw1);
  pack_w<32, 48><<<(3 * 27 * 256 + 255) / 256, 256, 0, stream>>>(W2, pw2);
  pack_w<48, 64><<<(4 * 41 * 256 + 255) / 256, 256, 0, stream>>>(W3, pw3);

  // conv layers: grid = (Mtiles/4, Ntiles, patches), 4 waves/block, dyn-LDS = KT KB
  conv_wmma<16, 128, 128, 2, 16><<<dim3(2048 / 4, 1, NPATCH), 128, 2  * 1024, stream>>>(x0, pw0, b0, a0);
  conv_wmma<8,  64,  64, 16, 32><<<dim3(256 / 4,  2, NPATCH), 128, 14 * 1024, stream>>>(a0, pw1, b1, a1);
  conv_wmma<4,  32,  32, 32, 48><<<dim3(32 / 4,   3, NPATCH), 128, 27 * 1024, stream>>>(a1, pw2, b2, a2);
  conv_wmma<2,  16,  16, 48, 64><<<dim3(4 / 4,    4, NPATCH), 128, 41 * 1024, stream>>>(a2, pw3, b3, a3);

  head<<<(NPATCH * 64 + 255) / 256, 256, 0, stream>>>(a3, Wo, bo, sig);
  scatter<<<(NPATCH * perPatch + 255) / 256, 256, 0, stream>>>(lab, focus, sig, ret);
}
